// LinearAttention_38448547234548
// MI455X (gfx1250) — compile-verified, matrix operand order checked by reading, not run on hardware
//
#include <hip/hip_runtime.h>
#include <hip/hip_bf16.h>

typedef __attribute__((ext_vector_type(16))) __bf16 v16bf;
typedef __attribute__((ext_vector_type(8)))  __bf16 v8bf;
typedef __attribute__((ext_vector_type(8)))  float  v8f;
typedef __attribute__((ext_vector_type(4)))  int    v4i;

#if __has_builtin(__builtin_amdgcn_global_load_async_to_lds_b128) && \
    __has_builtin(__builtin_amdgcn_s_wait_asynccnt)
#define USE_ASYNC_LDS 1
typedef __attribute__((address_space(1))) v4i* gptr_v4i;
typedef __attribute__((address_space(3))) v4i* lptr_v4i;
#else
#define USE_ASYNC_LDS 0
#endif

#define EPSF 1e-6f

static constexpr int kB = 8, kN = 4096, kD = 512, kH = 8, kd = 64;
static constexpr int kM = kB * kN;   // 32768 tokens
static constexpr int LDSW = 40;      // padded LDS row stride (bf16): banks r*20%64 all distinct

__device__ __forceinline__ v16bf pack16(v8bf lo, v8bf hi) {
  v16bf o;
#pragma unroll
  for (int i = 0; i < 8; ++i) { o[i] = lo[i]; o[8 + i] = hi[i]; }
  return o;
}

// WMMA 16x32 bf16 operand fetch from a K-contiguous matrix (row-major-K).
// ISA layout: lane<16 -> row=lane, K={0..7}U{16..23}; lane>=16 -> row=lane-16,
// K={8..15}U{24..31}. Two 16B loads per lane.
__device__ __forceinline__ v16bf ld_operand(const __bf16* __restrict__ base,
                                            int ld, int rc0, int kb) {
  const int lane = threadIdx.x & 31;
  const int r = lane & 15, sel = lane >> 4;
  const __bf16* p = base + (size_t)(rc0 + r) * (size_t)ld + (kb + sel * 8);
  return pack16(*(const v8bf*)p, *(const v8bf*)(p + 16));
}

// Same fetch from the block's LDS-staged 64xLDSW W-slab.
__device__ __forceinline__ v16bf ld_operand_lds(const __bf16* slab, int rc0) {
  const int lane = threadIdx.x & 31;
  const int r = lane & 15, sel = lane >> 4;
  const __bf16* p = slab + (rc0 + r) * LDSW + sel * 8;
  return pack16(*(const v8bf*)p, *(const v8bf*)(p + 16));
}

// Block of 128 threads (4 waves) computes a 128(M) x 64(N) tile of C = A*W^T.
// Per wave: 32x64 strip (2 M-tiles x 4 N-tiles). W k-slab (64x32) staged in LDS
// once per K-step via async-to-LDS when available.
__device__ __forceinline__ void gemm_block_128x64(
    const __bf16* __restrict__ A, int lda, int row0,
    const __bf16* __restrict__ Wg, int ldw, int col0,
    int K, v8f acc[2][4], __bf16* wslab) {
  const int crow = threadIdx.x >> 1;        // 0..63: W row within slab
  const int coff = (threadIdx.x & 1) * 16;  // 0 or 16 bf16 within the 32-wide slab row
  for (int kb = 0; kb < K; kb += 32) {
    const __bf16* gsrc = Wg + (size_t)(col0 + crow) * (size_t)ldw + kb + coff;
    __bf16* ldst = wslab + crow * LDSW + coff;
#if USE_ASYNC_LDS
    __builtin_amdgcn_global_load_async_to_lds_b128(
        (gptr_v4i)gsrc, (lptr_v4i)ldst, 0, 0);
    __builtin_amdgcn_global_load_async_to_lds_b128(
        (gptr_v4i)(gsrc + 8), (lptr_v4i)(ldst + 8), 0, 0);
    __builtin_amdgcn_s_wait_asynccnt(0);
#else
    *(v8bf*)ldst = *(const v8bf*)gsrc;
    *(v8bf*)(ldst + 8) = *(const v8bf*)(gsrc + 8);
#endif
    __syncthreads();
    v16bf a0 = ld_operand(A, lda, row0, kb);
    v16bf a1 = ld_operand(A, lda, row0 + 16, kb);
#pragma unroll
    for (int t = 0; t < 4; ++t) {
      v16bf b = ld_operand_lds(wslab, t * 16);
      acc[0][t] = __builtin_amdgcn_wmma_f32_16x16x32_bf16(
          false, a0, false, b, (short)0, acc[0][t], false, false);
      acc[1][t] = __builtin_amdgcn_wmma_f32_16x16x32_bf16(
          false, a1, false, b, (short)0, acc[1][t], false, false);
    }
    __syncthreads();
  }
}

// 16x64-per-wave variant, operands straight from global (small GEMMs).
__device__ __forceinline__ void gemm_strip(const __bf16* __restrict__ A, int lda, int row0,
                                           const __bf16* __restrict__ W, int ldw, int col0,
                                           int K, v8f acc[4]) {
  for (int kb = 0; kb < K; kb += 32) {
    v16bf a = ld_operand(A, lda, row0, kb);
#pragma unroll
    for (int t = 0; t < 4; ++t) {
      v16bf b = ld_operand(W, ldw, col0 + t * 16, kb);
      acc[t] = __builtin_amdgcn_wmma_f32_16x16x32_bf16(
          false, a, false, b, (short)0, acc[t], false, false);
    }
  }
}

__global__ __launch_bounds__(256)
void k_cvt_bf16(const float* __restrict__ in, __bf16* __restrict__ out, int n) {
  int i = blockIdx.x * blockDim.x + threadIdx.x;
  const int stride = gridDim.x * blockDim.x;
  for (; i < n; i += stride) out[i] = (__bf16)in[i];
}

// QKV projection: qkv[m, j] = sum_k x[m,k] * w_qkv[j,k]; epilogue applies
// elu+1+eps to q,k and scatters into WMMA-friendly layouts.
__global__ __launch_bounds__(128)
void k_qkv(const __bf16* __restrict__ xb, const __bf16* __restrict__ wq,
           __bf16* __restrict__ qb, __bf16* __restrict__ kT, __bf16* __restrict__ vT) {
  __shared__ __bf16 wslab[64 * LDSW];
  const int wave = threadIdx.x >> 5;
  const int lane = threadIdx.x & 31;
  const int sel = lane >> 4, nloc = lane & 15;
  const int row0 = blockIdx.x * 128 + wave * 32;
  const int col0 = blockIdx.y * 64;
  const int which = blockIdx.y >> 3;  // 0=q, 1=k, 2=v
  const int h = blockIdx.y & 7;
  v8f acc[2][4] = {};
  gemm_block_128x64(xb, kD, row0, wq, kD, col0, kD, acc, wslab);
#pragma unroll
  for (int mi = 0; mi < 2; ++mi) {
#pragma unroll
    for (int t = 0; t < 4; ++t) {
      const int dd = t * 16 + nloc;
#pragma unroll
      for (int r = 0; r < 8; ++r) {
        const int m = row0 + mi * 16 + r + 8 * sel;
        const int b = m >> 12, n = m & (kN - 1);
        float v = acc[mi][t][r];
        if (which < 2) v = (v < 0.f) ? (__expf(v) + EPSF) : (v + 1.f + EPSF);
        const size_t bh = (size_t)(b * kH + h);
        if (which == 0)      qb[(bh * kN + n) * kd + dd] = (__bf16)v;   // [B,H,N,d]
        else if (which == 1) kT[(bh * kd + dd) * kN + n] = (__bf16)v;   // [B,H,d,N]
        else                 vT[(bh * kd + dd) * kN + n] = (__bf16)v;   // [B,H,d,N]
      }
    }
  }
}

// kv[dd,e] = sum_n kT[dd,n]*vT[e,n] per (b,h); store transposed bf16 (kvT[e,dd])
// as the W-operand of the q@kv GEMM. Also ksum[dd] = sum_n kT[dd,n].
__global__ __launch_bounds__(128)
void k_kv(const __bf16* __restrict__ kT, const __bf16* __restrict__ vT,
          __bf16* __restrict__ kvT, float* __restrict__ ksum) {
  const int bh = blockIdx.x;
  const __bf16* A = kT + (size_t)bh * kd * kN;
  const __bf16* W = vT + (size_t)bh * kd * kN;
  const int wave = threadIdx.x >> 5;
  const int lane = threadIdx.x & 31;
  const int sel = lane >> 4, nloc = lane & 15;
  v8f acc[4] = {};
  gemm_strip(A, kN, wave * 16, W, kN, 0, kN, acc);
#pragma unroll
  for (int t = 0; t < 4; ++t) {
    const int e = t * 16 + nloc;
#pragma unroll
    for (int r = 0; r < 8; ++r) {
      const int dd = wave * 16 + r + 8 * sel;
      kvT[(size_t)bh * (kd * kd) + (size_t)e * kd + dd] = (__bf16)acc[t][r];
    }
  }
  if (threadIdx.x < kd) {  // 64 row-sums, fp32 accumulate
    const v8bf* pr = (const v8bf*)(A + (size_t)threadIdx.x * kN);
    float s = 0.f;
    for (int i = 0; i < kN / 8; ++i) {
      v8bf v = pr[i];
#pragma unroll
      for (int j = 0; j < 8; ++j) s += (float)v[j];
    }
    ksum[bh * kd + threadIdx.x] = s;
  }
}

// attn[n,e] = (q[n,:] @ kv[:,e]) / (q[n,:].ksum + eps); write [B,N,D] bf16.
__global__ __launch_bounds__(128)
void k_attn(const __bf16* __restrict__ qb, const __bf16* __restrict__ kvT,
            const float* __restrict__ ksum, __bf16* __restrict__ attn) {
  const int bh = blockIdx.x;
  const int n0 = blockIdx.y * 64;
  const int b = bh >> 3, h = bh & 7;
  __shared__ float denomS[64];
  const __bf16* Q = qb + (size_t)bh * kN * kd;
  if (threadIdx.x < 64) {
    const __bf16* qr = Q + (size_t)(n0 + threadIdx.x) * kd;
    const float* ks = ksum + bh * kd;
    float s = 0.f;
#pragma unroll 8
    for (int i = 0; i < kd; ++i) s += (float)qr[i] * ks[i];
    denomS[threadIdx.x] = s + EPSF;
  }
  __syncthreads();
  const int wave = threadIdx.x >> 5;
  const int lane = threadIdx.x & 31;
  const int sel = lane >> 4, nloc = lane & 15;
  v8f acc[4] = {};
  gemm_strip(Q, kd, n0 + wave * 16, kvT + (size_t)bh * kd * kd, kd, 0, kd, acc);
#pragma unroll
  for (int t = 0; t < 4; ++t) {
    const int e = t * 16 + nloc;
#pragma unroll
    for (int r = 0; r < 8; ++r) {
      const int rl = wave * 16 + r + 8 * sel;
      const float v = acc[t][r] / denomS[rl];
      attn[((size_t)b * kN + (n0 + rl)) * kD + h * kd + e] = (__bf16)v;
    }
  }
}

// out = attn @ w_proj^T + b_proj (fp32 out)
__global__ __launch_bounds__(128)
void k_proj(const __bf16* __restrict__ attn, const __bf16* __restrict__ wp,
            const float* __restrict__ bias, float* __restrict__ out) {
  __shared__ __bf16 wslab[64 * LDSW];
  const int wave = threadIdx.x >> 5;
  const int lane = threadIdx.x & 31;
  const int sel = lane >> 4, nloc = lane & 15;
  const int row0 = blockIdx.x * 128 + wave * 32;
  const int col0 = blockIdx.y * 64;
  v8f acc[2][4] = {};
  gemm_block_128x64(attn, kD, row0, wp, kD, col0, kD, acc, wslab);
#pragma unroll
  for (int mi = 0; mi < 2; ++mi) {
#pragma unroll
    for (int t = 0; t < 4; ++t) {
      const int c = col0 + t * 16 + nloc;
      const float bb = bias[c];
#pragma unroll
      for (int r = 0; r < 8; ++r) {
        const int m = row0 + mi * 16 + r + 8 * sel;
        out[(size_t)m * kD + c] = acc[mi][t][r] + bb;
      }
    }
  }
}

extern "C" void kernel_launch(void* const* d_in, const int* in_sizes, int n_in,
                              void* d_out, int out_size, void* d_ws, size_t ws_size,
                              hipStream_t stream) {
  (void)in_sizes; (void)n_in; (void)out_size; (void)ws_size;
  const float* x      = (const float*)d_in[0];
  const float* w_qkv  = (const float*)d_in[1];
  const float* w_proj = (const float*)d_in[2];
  const float* b_proj = (const float*)d_in[3];
  // d_in[4] = heads (8), baked into constants.

  char* ws = (char*)d_ws;
  size_t off = 0;
  auto alloc = [&](size_t bytes) -> void* {
    void* p = ws + off;
    off = (off + bytes + 255) & ~(size_t)255;
    return p;
  };
  __bf16* xb   = (__bf16*)alloc((size_t)kM * kD * 2);          // 32 MB
  __bf16* wqb  = (__bf16*)alloc((size_t)3 * kD * kD * 2);      // 1.5 MB
  __bf16* wpb  = (__bf16*)alloc((size_t)kD * kD * 2);          // 0.5 MB
  __bf16* qb   = (__bf16*)alloc((size_t)kM * kD * 2);          // 32 MB
  __bf16* kT   = (__bf16*)alloc((size_t)kM * kD * 2);          // 32 MB
  __bf16* vT   = (__bf16*)alloc((size_t)kM * kD * 2);          // 32 MB
  __bf16* kvT  = (__bf16*)alloc((size_t)kB * kH * kd * kd * 2);
  float*  ksum = (float*)alloc((size_t)kB * kH * kd * 4);
  __bf16* attn = (__bf16*)alloc((size_t)kM * kD * 2);          // 32 MB

  k_cvt_bf16<<<dim3(4096), dim3(256), 0, stream>>>(x, xb, kM * kD);
  k_cvt_bf16<<<dim3(1024), dim3(256), 0, stream>>>(w_qkv, wqb, 3 * kD * kD);
  k_cvt_bf16<<<dim3(512),  dim3(256), 0, stream>>>(w_proj, wpb, kD * kD);

  k_qkv <<<dim3(kM / 128, (3 * kD) / 64), dim3(128), 0, stream>>>(xb, wqb, qb, kT, vT);
  k_kv  <<<dim3(kB * kH),                 dim3(128), 0, stream>>>(kT, vT, kvT, ksum);
  k_attn<<<dim3(kB * kH, kN / 64),        dim3(128), 0, stream>>>(qb, kvT, ksum, attn);
  k_proj<<<dim3(kM / 128, kD / 64),       dim3(128), 0, stream>>>(attn, wpb, b_proj,
                                                                  (float*)d_out);
}
